// TransformerMHSA_57200374448292
// MI455X (gfx1250) — matrix operand order repaired and checked by reading, not compiled, not run on hardware
//
#include <hip/hip_runtime.h>

// ---- problem constants (match reference) ----
#define NN 1024    // sequence length
#define BB 4       // batch
#define DD 1024    // model dim
#define HH 16      // heads
#define HD 64      // head dim
#define MM (BB*NN) // 4096 GEMM rows

typedef __attribute__((ext_vector_type(16))) _Float16 v16h;
typedef __attribute__((ext_vector_type(8)))  _Float16 v8h;
typedef __attribute__((ext_vector_type(8)))  float    v8f;

// ---------------- WMMA helpers (gfx1250 layouts) ----------------

__device__ __forceinline__ v16h pack16(v8h lo, v8h hi) {
  v16h r;
#pragma unroll
  for (int i = 0; i < 8; ++i) { r[i] = lo[i]; r[i + 8] = hi[i]; }
  return r;
}

// A-matrix fragment, 16(M) x 32(K) f16, from row-major [row][ld] memory.
__device__ __forceinline__ v16h load_a(const _Float16* base, int ld, int k0) {
  int lane = threadIdx.x & 31;
  const _Float16* p = base + (size_t)(lane & 15) * ld + k0 + ((lane >> 4) << 3);
  return pack16(*(const v8h*)p, *(const v8h*)(p + 16));
}

// B-matrix fragment, 32(K) x 16(N) f16; row n of `base` = column n (contig K).
__device__ __forceinline__ v16h load_b(const _Float16* base, int ld, int k0) {
  int lane = threadIdx.x & 31;
  const _Float16* p = base + (size_t)(lane & 15) * ld + k0 + ((lane >> 4) << 3);
  return pack16(*(const v8h*)p, *(const v8h*)(p + 16));
}

__device__ __forceinline__ v8f wmma16(v16h a, v16h b, v8f c) {
  return __builtin_amdgcn_wmma_f32_16x16x32_f16(false, a, false, b,
                                                (short)0, c, false, false);
}

// gfx1250 async copy: 64 bytes global -> LDS (4x b128), tracked by ASYNCcnt.
// INST_OFFSET applies to both the LDS and global address (ISA 08 §4.4).
__device__ __forceinline__ void async_copy64(unsigned lds, const _Float16* g) {
  unsigned long long ga = (unsigned long long)(uintptr_t)g;
  asm volatile(
      "global_load_async_to_lds_b128 %0, %1, off\n\t"
      "global_load_async_to_lds_b128 %0, %1, off offset:16\n\t"
      "global_load_async_to_lds_b128 %0, %1, off offset:32\n\t"
      "global_load_async_to_lds_b128 %0, %1, off offset:48"
      :: "v"(lds), "v"(ga) : "memory");
}
__device__ __forceinline__ void wait_async0() {
  asm volatile("s_wait_asynccnt 0x0" ::: "memory");
}
// generic-pointer low 32 bits == LDS byte offset (ISA 10.2 aperture mapping)
__device__ __forceinline__ unsigned lds_off(const void* p) {
  return (unsigned)(uintptr_t)p;
}

// ---------------- 1) LayerNorm (fp32 in -> f16 out, row = b*N + n) --------

__global__ __launch_bounds__(256) void ln_kernel(
    const float* __restrict__ x, const float* __restrict__ g,
    const float* __restrict__ bsv, _Float16* __restrict__ xln) {
  int wave = threadIdx.x >> 5, lane = threadIdx.x & 31;
  int r = blockIdx.x * 8 + wave;
  int b = r >> 10, n = r & (NN - 1);
  const float* row = x + ((size_t)n * BB + b) * DD;  // input is [N,B,D]
  float v[32], s = 0.f, ss = 0.f;
#pragma unroll
  for (int i = 0; i < 32; ++i) {
    float t = row[lane + i * 32];
    v[i] = t; s += t; ss += t * t;
  }
#pragma unroll
  for (int off = 16; off; off >>= 1) {
    s  += __shfl_xor(s,  off, 32);
    ss += __shfl_xor(ss, off, 32);
  }
  float mean = s * (1.f / DD);
  float var  = ss * (1.f / DD) - mean * mean;
  float rs   = rsqrtf(var + 1e-5f);
#pragma unroll
  for (int i = 0; i < 32; ++i) {
    int d = lane + i * 32;
    xln[(size_t)r * DD + d] = (_Float16)((v[i] - mean) * rs * g[d] + bsv[d]);
  }
}

// ---------------- 2) weight downconvert ----------------

__global__ void cvt_f32_f16(const float* __restrict__ in,
                            _Float16* __restrict__ out, int n) {
  int i = blockIdx.x * blockDim.x + threadIdx.x;
  if (i < n) out[i] = (_Float16)in[i];
}

// ---------------- shared GEMM mainloop: 32x64 wave tile, 2-stage pipeline --

__device__ __forceinline__ void gemm_mainloop(
    const _Float16* __restrict__ arow,   // rows m0..m0+31, row-major [*, DD]
    const _Float16* __restrict__ wb,     // rows e0..e0+63 of W, row-major
    v8f (&c)[2][4]) {
  int lane = threadIdx.x & 31;
  const _Float16* pa = arow + (size_t)(lane & 15) * DD;
  v16h a0 = load_a(arow, DD, 0);
  v16h a1 = load_a(arow + (size_t)16 * DD, DD, 0);
  v16h b0 = load_b(wb, DD, 0);
  v16h b1 = load_b(wb + (size_t)16 * DD, DD, 0);
  v16h b2 = load_b(wb + (size_t)32 * DD, DD, 0);
  v16h b3 = load_b(wb + (size_t)48 * DD, DD, 0);
  for (int k0 = 32; k0 < DD; k0 += 32) {
    // next K-step's fragments in flight while current WMMAs execute
    __builtin_prefetch(pa + k0 + 224, 0, 0);
    v16h na0 = load_a(arow, DD, k0);
    v16h na1 = load_a(arow + (size_t)16 * DD, DD, k0);
    v16h nb0 = load_b(wb, DD, k0);
    v16h nb1 = load_b(wb + (size_t)16 * DD, DD, k0);
    v16h nb2 = load_b(wb + (size_t)32 * DD, DD, k0);
    v16h nb3 = load_b(wb + (size_t)48 * DD, DD, k0);
    c[0][0] = wmma16(a0, b0, c[0][0]); c[1][0] = wmma16(a1, b0, c[1][0]);
    c[0][1] = wmma16(a0, b1, c[0][1]); c[1][1] = wmma16(a1, b1, c[1][1]);
    c[0][2] = wmma16(a0, b2, c[0][2]); c[1][2] = wmma16(a1, b2, c[1][2]);
    c[0][3] = wmma16(a0, b3, c[0][3]); c[1][3] = wmma16(a1, b3, c[1][3]);
    a0 = na0; a1 = na1; b0 = nb0; b1 = nb1; b2 = nb2; b3 = nb3;
  }
  c[0][0] = wmma16(a0, b0, c[0][0]); c[1][0] = wmma16(a1, b0, c[1][0]);
  c[0][1] = wmma16(a0, b1, c[0][1]); c[1][1] = wmma16(a1, b1, c[1][1]);
  c[0][2] = wmma16(a0, b2, c[0][2]); c[1][2] = wmma16(a1, b2, c[1][2]);
  c[0][3] = wmma16(a0, b3, c[0][3]); c[1][3] = wmma16(a1, b3, c[1][3]);
}

// ---------------- 3) QKV GEMM + scatter (V stored TRANSPOSED) -------------
// grid (MM/32, 3072/256), block 128; wave -> 32x64 tile.

__global__ __launch_bounds__(128) void qkv_gemm(
    const _Float16* __restrict__ xln, const _Float16* __restrict__ w,
    _Float16* __restrict__ Qb, _Float16* __restrict__ Kb,
    _Float16* __restrict__ VbT) {
  int wave = threadIdx.x >> 5, lane = threadIdx.x & 31;
  int m0 = blockIdx.x * 32;
  int e0 = blockIdx.y * 256 + wave * 64;
  v8f c[2][4] = {};
  gemm_mainloop(xln + (size_t)m0 * DD, w + (size_t)e0 * DD, c);

  int half = lane >> 4, col16 = lane & 15;
#pragma unroll
  for (int t = 0; t < 4; ++t) {
    int col = e0 + t * 16 + col16;
    int reg = col >> 10, ec = col & (DD - 1);
    int h = ec >> 6, hd = ec & (HD - 1);
    float scale = (reg == 0) ? 0.125f : 1.0f;  // 1/sqrt(HD)
#pragma unroll
    for (int gidx = 0; gidx < 2; ++gidx)
#pragma unroll
      for (int r = 0; r < 8; ++r) {
        int rowm = m0 + gidx * 16 + r + half * 8;  // = b*N + n
        int bb = rowm >> 10, n = rowm & (NN - 1);
        _Float16 val = (_Float16)(c[gidx][t][r] * scale);
        size_t bh = (size_t)bb * HH + h;
        if (reg == 0)      Qb[(bh * NN + n) * HD + hd] = val;
        else if (reg == 1) Kb[(bh * NN + n) * HD + hd] = val;
        else               VbT[(bh * HD + hd) * NN + n] = val;  // transposed
      }
  }
}

// ---------------- 4) flash attention, async double-buffered staging -------
// grid (N/64, B*H), block 128; block owns 64 queries of one (b,h).

__global__ __launch_bounds__(128) void attn_kernel(
    const _Float16* __restrict__ Qb, const _Float16* __restrict__ Kb,
    const _Float16* __restrict__ VbT, const unsigned char* __restrict__ kpm,
    const unsigned char* __restrict__ amask, _Float16* __restrict__ Ob) {
  __shared__ __align__(16) _Float16 Klds[2][64][64];   // [buf][key][d]
  __shared__ __align__(16) _Float16 Vtl[2][64][64];    // [buf][d][key]
  __shared__ __align__(16) _Float16 Plds[4][16][64];   // wave-private P tile

  int wave = threadIdx.x >> 5, lane = threadIdx.x & 31;
  int bh = blockIdx.y, b = bh >> 4, h = bh & (HH - 1);
  int q0 = blockIdx.x * 64 + wave * 16;
  int half = lane >> 4, col16 = lane & 15;

  const _Float16* qbase  = Qb  + ((size_t)bh << 16);
  const _Float16* kbase  = Kb  + ((size_t)bh << 16);
  const _Float16* vtbase = VbT + ((size_t)bh << 16);

  v16h aq0 = load_a(qbase + (size_t)q0 * HD, HD, 0);
  v16h aq1 = load_a(qbase + (size_t)q0 * HD, HD, 32);

  v8f o[4] = {};
  float mi[8], li[8];
#pragma unroll
  for (int r = 0; r < 8; ++r) { mi[r] = -1e30f; li[r] = 0.f; }

  int row2 = threadIdx.x >> 1;          // key-row (K) / d-row (Vt)
  int h32  = (threadIdx.x & 1) * 32;    // 32-half slice -> 64B per thread
  // stage chunk 0 into buffer 0
  async_copy64(lds_off(&Klds[0][row2][h32]),
               kbase + (size_t)row2 * HD + h32);
  async_copy64(lds_off(&Vtl[0][row2][h32]),
               vtbase + (size_t)row2 * NN + h32);

  int nch = blockIdx.x + 1;  // causal: up to (incl.) diagonal chunk
  for (int cc = 0; cc < nch; ++cc) {
    int buf = cc & 1;
    wait_async0();
    __syncthreads();
    if (cc + 1 < nch) {  // prefetch next chunk into other buffer
      int k1 = (cc + 1) * 64;
      async_copy64(lds_off(&Klds[buf ^ 1][row2][h32]),
                   kbase + (size_t)(k1 + row2) * HD + h32);
      async_copy64(lds_off(&Vtl[buf ^ 1][row2][h32]),
                   vtbase + (size_t)row2 * NN + k1 + h32);
    }
    int k0 = cc * 64;

    // S = Q * K^T
    v8f s[4] = {};
#pragma unroll
    for (int t = 0; t < 4; ++t) {
      s[t] = wmma16(aq0, load_b(&Klds[buf][t * 16][0], 64, 0), s[t]);
      s[t] = wmma16(aq1, load_b(&Klds[buf][t * 16][0], 64, 32), s[t]);
    }
    // masks (key padding + attn mask from memory)
#pragma unroll
    for (int t = 0; t < 4; ++t) {
      int col = k0 + t * 16 + col16;
      bool kv = kpm[b * NN + col] != 0;
#pragma unroll
      for (int r = 0; r < 8; ++r) {
        int row = q0 + r + half * 8;
        bool vis = kv && (amask[(size_t)row * NN + col] != 0);
        if (!vis) s[t][r] = -1e30f;
      }
    }
    // online softmax (row groups = 16-lane halves of the C layout)
#pragma unroll
    for (int r = 0; r < 8; ++r) {
      float mloc = fmaxf(fmaxf(s[0][r], s[1][r]), fmaxf(s[2][r], s[3][r]));
      mloc = fmaxf(mloc, __shfl_xor(mloc, 1, 32));
      mloc = fmaxf(mloc, __shfl_xor(mloc, 2, 32));
      mloc = fmaxf(mloc, __shfl_xor(mloc, 4, 32));
      mloc = fmaxf(mloc, __shfl_xor(mloc, 8, 32));
      float mn = fmaxf(mi[r], mloc);
      float alpha = __expf(mi[r] - mn);
      float ps = 0.f;
#pragma unroll
      for (int t = 0; t < 4; ++t) {
        float p = __expf(s[t][r] - mn);
        s[t][r] = p; ps += p;
      }
      ps += __shfl_xor(ps, 1, 32); ps += __shfl_xor(ps, 2, 32);
      ps += __shfl_xor(ps, 4, 32); ps += __shfl_xor(ps, 8, 32);
      li[r] = li[r] * alpha + ps;
      mi[r] = mn;
#pragma unroll
      for (int t = 0; t < 4; ++t) o[t][r] *= alpha;
    }
    // relayout P: C-layout -> A-layout through wave-private LDS
#pragma unroll
    for (int t = 0; t < 4; ++t)
#pragma unroll
      for (int r = 0; r < 8; ++r)
        Plds[wave][r + half * 8][t * 16 + col16] = (_Float16)s[t][r];
    asm volatile("s_wait_dscnt 0x0" ::: "memory");

    v16h ap0 = load_a(&Plds[wave][0][0], 64, 0);
    v16h ap1 = load_a(&Plds[wave][0][0], 64, 32);
#pragma unroll
    for (int t = 0; t < 4; ++t) {  // O += P * V
      o[t] = wmma16(ap0, load_b(&Vtl[buf][t * 16][0], 64, 0), o[t]);
      o[t] = wmma16(ap1, load_b(&Vtl[buf][t * 16][0], 64, 32), o[t]);
    }
  }
  // epilogue: normalize, write f16 row-major [n*B+b][D]
#pragma unroll
  for (int t = 0; t < 4; ++t)
#pragma unroll
    for (int r = 0; r < 8; ++r) {
      int row = q0 + r + half * 8;  // query index n
      int d = h * HD + t * 16 + col16;
      Ob[((size_t)row * BB + b) * DD + d] = (_Float16)(o[t][r] / li[r]);
    }
}

// ---------------- 5) output GEMM: out[m,e] = sum_d Ob[m,d]*Wout[e,d] ------

__global__ __launch_bounds__(128) void out_gemm(
    const _Float16* __restrict__ ob, const _Float16* __restrict__ w,
    float* __restrict__ out) {
  int wave = threadIdx.x >> 5, lane = threadIdx.x & 31;
  int m0 = blockIdx.x * 32;
  int e0 = blockIdx.y * 256 + wave * 64;
  v8f c[2][4] = {};
  gemm_mainloop(ob + (size_t)m0 * DD, w + (size_t)e0 * DD, c);
  int half = lane >> 4, col16 = lane & 15;
#pragma unroll
  for (int t = 0; t < 4; ++t) {
    int col = e0 + t * 16 + col16;
#pragma unroll
    for (int gidx = 0; gidx < 2; ++gidx)
#pragma unroll
      for (int r = 0; r < 8; ++r)
        out[(size_t)(m0 + gidx * 16 + r + half * 8) * DD + col] =
            c[gidx][t][r];
  }
}

// ---------------- launch ----------------

extern "C" void kernel_launch(void* const* d_in, const int* in_sizes, int n_in,
                              void* d_out, int out_size, void* d_ws,
                              size_t ws_size, hipStream_t stream) {
  const float* input = (const float*)d_in[0];
  const unsigned char* kpm = (const unsigned char*)d_in[1];    // bool [B,N]
  const unsigned char* amask = (const unsigned char*)d_in[2];  // bool [N,N]
  const float* g = (const float*)d_in[3];
  const float* bsv = (const float*)d_in[4];
  const float* wqkv = (const float*)d_in[5];  // [3D, D]
  const float* wout = (const float*)d_in[6];  // [D, D]
  float* out = (float*)d_out;

  char* ws = (char*)d_ws;
  _Float16* xln = (_Float16*)(ws);                  // 8 MB [M, D]
  _Float16* wqh = (_Float16*)(ws + (8ull << 20));   // 6 MB [3D, D]
  _Float16* woh = (_Float16*)(ws + (14ull << 20));  // 2 MB [D, D]
  _Float16* Qb  = (_Float16*)(ws + (16ull << 20));  // 8 MB [B,H,N,HD]
  _Float16* Kb  = (_Float16*)(ws + (24ull << 20));  // 8 MB [B,H,N,HD]
  _Float16* VbT = (_Float16*)(ws + (32ull << 20));  // 8 MB [B,H,HD,N]
  _Float16* Ob  = (_Float16*)(ws + (40ull << 20));  // 8 MB [M, D]

  ln_kernel<<<MM / 8, 256, 0, stream>>>(input, g, bsv, xln);
  cvt_f32_f16<<<(3 * DD * DD) / 256, 256, 0, stream>>>(wqkv, wqh, 3 * DD * DD);
  cvt_f32_f16<<<(DD * DD) / 256, 256, 0, stream>>>(wout, woh, DD * DD);
  qkv_gemm<<<dim3(MM / 32, (3 * DD) / 256), 128, 0, stream>>>(xln, wqh, Qb, Kb,
                                                              VbT);
  attn_kernel<<<dim3(NN / 64, BB * HH), 128, 0, stream>>>(Qb, Kb, VbT, kpm,
                                                          amask, Ob);
  out_gemm<<<dim3(MM / 32, DD / 256), 128, 0, stream>>>(Ob, woh, out);
}
